// AugmentedGeometryScaledDotProductAttention_54683523612802
// MI455X (gfx1250) — matrix operand order repaired and checked by reading, not compiled, not run on hardware
//
#include <hip/hip_runtime.h>
#include <hip/hip_bf16.h>

#define B_  4
#define H_  16
#define NQ_ 1024
#define NK_ 1024
#define DM_ 1024
#define DK_ 64

typedef __attribute__((ext_vector_type(16))) __bf16 v16bf;
typedef __attribute__((ext_vector_type(8)))  float  v8f;
typedef __attribute__((ext_vector_type(4)))  unsigned int v4u;
typedef __attribute__((ext_vector_type(8)))  int v8i_;
typedef __attribute__((ext_vector_type(4)))  int v4i_;

#if __has_builtin(__builtin_amdgcn_tensor_load_to_lds)
#define USE_TDM 1
#else
#define USE_TDM 0
#endif

__device__ __forceinline__ v8f wmma_bf16(v16bf a, v16bf b, v8f c) {
  // (neg_a, A, neg_b, B, c_mod, C, reuse_a, reuse_b)
  return __builtin_amdgcn_wmma_f32_16x16x32_bf16(false, a, false, b, (short)0, c,
                                                 false, false);
}

// A-fragment (16x32 bf16): this lane owns row m = lane&15, group g = lane>>4.
// Element e holds K = e + 16*g.  rowp points at [m][kbase].
__device__ __forceinline__ v16bf a_frag(const __bf16* rowp, int g) {
  v16bf a;
#pragma unroll
  for (int e = 0; e < 16; ++e) {
    int k = e + 16 * g;
    a[e] = rowp[k];
  }
  return a;
}

// Same as a_frag but reading raw bf16 bits from global workspace.
__device__ __forceinline__ v16bf a_frag_g(const unsigned short* rowp, int g) {
  v16bf a;
#pragma unroll
  for (int e = 0; e < 16; ++e) {
    int k = e + 16 * g;
    a[e] = __builtin_bit_cast(__bf16, rowp[k]);
  }
  return a;
}

// B-fragment (32x16 bf16): lane owns column n = lane&15, group kg = lane>>4.
// Element e holds K = e + 16*kg.  colp points at [n][kbase] (contiguous in K).
__device__ __forceinline__ v16bf b_frag(const __bf16* colp, int kg) {
  v16bf b;
#pragma unroll
  for (int e = 0; e < 16; ++e) b[e] = colp[e + 16 * kg];
  return b;
}

// B-fragment gathered from a row-major tile: element e at p[e*stride].
__device__ __forceinline__ v16bf b_frag_strided(const __bf16* p, int stride) {
  v16bf b;
#pragma unroll
  for (int e = 0; e < 16; ++e) b[e] = p[e * stride];
  return b;
}

#if USE_TDM
// Issue a TDM 2D tile load of bf16 data: tile_d0 elements per row (contiguous
// in memory), `rows` rows with `row_stride` elements between them, into LDS at
// byte offset lds_off.  pad_code: +1 dword of LDS padding every 2^(pad_code+1)
// dwords (matches our [.. ][d0+2] LDS arrays).  D# layout per ISA 08 §8.
__device__ __forceinline__ void tdm_load_2d_bf16(unsigned lds_off,
                                                 const unsigned short* gptr,
                                                 unsigned tile_d0, unsigned rows,
                                                 unsigned long long row_stride,
                                                 unsigned pad_code) {
  unsigned long long ga = (unsigned long long)(size_t)gptr;
  v4u g0;
  g0[0] = 1u;                                   // count=1, user descriptor
  g0[1] = lds_off;                              // lds_addr (bytes)
  g0[2] = (unsigned)(ga & 0xFFFFFFFFu);         // global_addr[31:0]
  g0[3] = (unsigned)((ga >> 32) & 0x01FFFFFFu)  // global_addr[56:32]
          | (2u << 30);                         // type = 2 ("image")
  v8i_ g1;
  unsigned w0 = (1u << 16)                      // data_size = 2 bytes
                | (1u << 20)                    // pad_enable
                | (pad_code << 22);             // pad_interval; pad_amount=0 (1 dw)
  g1[0] = (int)w0;
  g1[1] = (int)((tile_d0 & 0xFFFFu) << 16);     // tensor_dim0[15:0]
  g1[2] = (int)(((tile_d0 >> 16) & 0xFFFFu) | ((rows & 0xFFFFu) << 16));
  g1[3] = (int)(((rows >> 16) & 0xFFFFu) | ((tile_d0 & 0xFFFFu) << 16)); // tile_dim0
  g1[4] = (int)(rows & 0xFFFFu);                // tile_dim1; tile_dim2 = 0
  g1[5] = (int)(unsigned)(row_stride & 0xFFFFFFFFu);     // tensor_dim0_stride lo
  g1[6] = (int)(unsigned)((row_stride >> 32) & 0xFFFFu); // stride hi; dim1_stride=0
  g1[7] = 0;
  v4i_ z = {};
#if __has_include(<hip/amd_detail/amd_gfx1250_TDM.h>)
  v8i_ zb = {};
  __builtin_amdgcn_tensor_load_to_lds(g0, g1, z, z, zb, 0);  // clang-23 form
#else
  __builtin_amdgcn_tensor_load_to_lds(g0, g1, z, z, 0);      // ROCm 7.2 form
#endif
}
#endif

// ---------------------------------------------------------------------------
// Kernel 1: fused Q/K/V projection.  64x64 output tile per block, 4 waves,
// each wave owns 16 rows.  Output written head-major (b,h,n,d) as bf16.
// blockIdx.z: 0=Q (scaled by 1/sqrt(DK)), 1=K, 2=V.  f32->bf16 conversion
// keeps this staging on the VALU path (TDM cannot convert).
// ---------------------------------------------------------------------------
__global__ __launch_bounds__(128) void qkv_proj_kernel(
    const float* __restrict__ Xq, const float* __restrict__ Xk,
    const float* __restrict__ Xv, const float* __restrict__ Wq,
    const float* __restrict__ Wk, const float* __restrict__ Wv,
    const float* __restrict__ bq, const float* __restrict__ bk,
    const float* __restrict__ bv, unsigned short* __restrict__ qh,
    unsigned short* __restrict__ kh, unsigned short* __restrict__ vh) {
  __shared__ __bf16 sA[64][34];   // [row][k]  (activation tile)
  __shared__ __bf16 sBt[64][34];  // [col][k]  (weight tile, transposed)

  const int tid = threadIdx.x;
  const int wave = tid >> 5;
  const int lane = tid & 31;
  const int ln = lane & 15, lg = lane >> 4;
  const int col0 = blockIdx.x * 64;
  const int row0 = blockIdx.y * 64;

  const float* X;
  const float* W;
  const float* bias;
  unsigned short* outp;
  float scale;
  if (blockIdx.z == 0) { X = Xq; W = Wq; bias = bq; outp = qh; scale = 0.125f; }
  else if (blockIdx.z == 1) { X = Xk; W = Wk; bias = bk; outp = kh; scale = 1.0f; }
  else { X = Xv; W = Wv; bias = bv; outp = vh; scale = 1.0f; }

  v8f acc[4] = {};
  for (int k0 = 0; k0 < DM_; k0 += 32) {
    for (int t = tid; t < 64 * 32; t += 128) {
      int r = t >> 5, c = t & 31;
      sA[r][c] = (__bf16)X[(size_t)(row0 + r) * DM_ + k0 + c];
    }
    for (int t = tid; t < 32 * 64; t += 128) {
      int kk = t >> 6, c = t & 63;
      sBt[c][kk] = (__bf16)W[(size_t)(k0 + kk) * DM_ + col0 + c];
    }
    __syncthreads();
    v16bf a = a_frag(&sA[wave * 16 + ln][0], lg);
#pragma unroll
    for (int nt = 0; nt < 4; ++nt) {
      v16bf b = b_frag(&sBt[nt * 16 + ln][0], lg);
      acc[nt] = wmma_bf16(a, b, acc[nt]);
    }
    __syncthreads();
  }
#pragma unroll
  for (int nt = 0; nt < 4; ++nt) {
#pragma unroll
    for (int i = 0; i < 8; ++i) {
      int m = lg * 8 + i;
      int n = col0 + nt * 16 + ln;
      float vv = (acc[nt][i] + bias[n]) * scale;
      int gr = row0 + wave * 16 + m;  // global row in [0, B*NQ)
      int bb = gr >> 10, nq = gr & 1023;
      int hh = n >> 6, dd = n & 63;
      outp[(size_t)((bb * H_ + hh) * NQ_ + nq) * DK_ + dd] =
          __builtin_bit_cast(unsigned short, (__bf16)vv);
    }
  }
}

// ---------------------------------------------------------------------------
// Kernel 2: flash attention with log-geometry bias.  One block per
// (64-query tile, head, batch); each of the 4 waves owns 16 query rows.
// K/V tiles are staged by the Tensor Data Mover (bf16 2D tile copies);
// wave 0 issues the descriptors and publishes completion via the barrier.
// ---------------------------------------------------------------------------
__global__ __launch_bounds__(128) void attn_kernel(
    const unsigned short* __restrict__ qh, const unsigned short* __restrict__ kh,
    const unsigned short* __restrict__ vh, const float* __restrict__ geo,
    const unsigned char* __restrict__ mask, unsigned short* __restrict__ ao) {
  __shared__ __bf16 sK[64][66];       // [key][d] — B frags for QK^T
  __shared__ __bf16 sV[64][66];       // [key][d] — B frags for PV (strided)
  __shared__ __bf16 sP[4][16][66];    // per-wave P tile (A frags for PV)

  const int tid = threadIdx.x;
  const int wave = tid >> 5;
  const int lane = tid & 31;
  const int ln = lane & 15, lg = lane >> 4;
  const int q0 = blockIdx.x * 64;
  const int h = blockIdx.y;
  const int b = blockIdx.z;
  const size_t headoff = (size_t)(b * H_ + h) * NQ_ * DK_;
  const int qrow0 = q0 + wave * 16;

  // Q A-fragments, K-dim 0..31 and 32..63 (scale already folded into qh)
  const unsigned short* qrow = qh + headoff + (size_t)(qrow0 + ln) * DK_;
  v16bf qa0 = a_frag_g(qrow, lg);
  v16bf qa1 = a_frag_g(qrow + 32, lg);

  v8f O[4] = {};
  float run_max[8], run_sum[8];
#pragma unroll
  for (int i = 0; i < 8; ++i) { run_max[i] = -1e30f; run_sum[i] = 0.0f; }

  const float* geo_base = geo + (size_t)(b * H_ + h) * NQ_ * NK_;
  const unsigned char* mrow = mask + (size_t)b * NK_;

  for (int kt = 0; kt < NK_; kt += 64) {
#if USE_TDM
    if (wave == 0) {
      // 64x64 bf16 tiles; row = 128 B = 32 dwords, +1 dword LDS pad -> code 4.
      tdm_load_2d_bf16((unsigned)(size_t)&sK[0][0],
                       kh + headoff + (size_t)kt * DK_, 64, 64, DK_, 4);
      tdm_load_2d_bf16((unsigned)(size_t)&sV[0][0],
                       vh + headoff + (size_t)kt * DK_, 64, 64, DK_, 4);
      __builtin_amdgcn_s_wait_tensorcnt(0);
    }
#else
    for (int t = tid; t < 64 * 64; t += 128) {
      int r = t >> 6, c = t & 63;
      sK[r][c] = __builtin_bit_cast(__bf16, kh[headoff + (size_t)(kt + r) * DK_ + c]);
      sV[r][c] = __builtin_bit_cast(__bf16, vh[headoff + (size_t)(kt + r) * DK_ + c]);
    }
#endif
    __syncthreads();

    v8f S[4];
#pragma unroll
    for (int nt = 0; nt < 4; ++nt) {
      v16bf bk0 = b_frag(&sK[nt * 16 + ln][0], lg);
      v16bf bk1 = b_frag(&sK[nt * 16 + ln][32], lg);
      v8f s = {};
      s = wmma_bf16(qa0, bk0, s);
      s = wmma_bf16(qa1, bk1, s);
      int ncol = kt + nt * 16 + ln;
      bool msk = mrow[ncol] != 0;
      const float* gcol = geo_base + ncol;
#pragma unroll
      for (int i = 0; i < 8; ++i) {
        int m = lg * 8 + i;
        const float* gp = gcol + (size_t)(qrow0 + m) * NK_;
        float g = *gp;
        if (nt == 0 && kt + 64 < NK_) __builtin_prefetch(gp + 64, 0, 0);
        float sv = s[i] + __logf(fmaxf(g, 1e-6f));
        s[i] = msk ? -1e30f : sv;
      }
      S[nt] = s;
    }

    // Tile row-max, cross-lane over the 16 columns held by the lane group.
    float alpha[8], psum[8];
#pragma unroll
    for (int i = 0; i < 8; ++i) {
      float mx = fmaxf(fmaxf(S[0][i], S[1][i]), fmaxf(S[2][i], S[3][i]));
#pragma unroll
      for (int off = 1; off < 16; off <<= 1)
        mx = fmaxf(mx, __shfl_xor(mx, off, 32));
      float nm = fmaxf(run_max[i], mx);
      alpha[i] = __expf(run_max[i] - nm);
      run_max[i] = nm;
      psum[i] = 0.0f;
    }
    // P = exp(S - max), stage to LDS as bf16 in natural (row, col) order.
#pragma unroll
    for (int nt = 0; nt < 4; ++nt) {
#pragma unroll
      for (int i = 0; i < 8; ++i) {
        float p = __expf(S[nt][i] - run_max[i]);
        psum[i] += p;
        sP[wave][lg * 8 + i][nt * 16 + ln] = (__bf16)p;
      }
    }
#pragma unroll
    for (int i = 0; i < 8; ++i) {
      float s = psum[i];
#pragma unroll
      for (int off = 1; off < 16; off <<= 1) s += __shfl_xor(s, off, 32);
      run_sum[i] = run_sum[i] * alpha[i] + s;
    }
#pragma unroll
    for (int nt = 0; nt < 4; ++nt)
#pragma unroll
      for (int i = 0; i < 8; ++i) O[nt][i] *= alpha[i];
    __syncthreads();  // sP visible across lanes

    // O += P x V  (two 32-deep K steps over the 64 keys of this tile).
    // V B-fragment: N = d (lane), K = key -> stride-66 walk down sV rows.
#pragma unroll
    for (int kk = 0; kk < 64; kk += 32) {
      v16bf pa = a_frag(&sP[wave][ln][kk], lg);
#pragma unroll
      for (int nt = 0; nt < 4; ++nt) {
        v16bf bv = b_frag_strided(&sV[kk + 16 * lg][nt * 16 + ln], 66);
        O[nt] = wmma_bf16(pa, bv, O[nt]);
      }
    }
    __syncthreads();  // before next tile overwrites sK/sV
  }

#pragma unroll
  for (int i = 0; i < 8; ++i) run_sum[i] = 1.0f / run_sum[i];
#pragma unroll
  for (int nt = 0; nt < 4; ++nt) {
#pragma unroll
    for (int i = 0; i < 8; ++i) {
      int m = lg * 8 + i;
      int q = qrow0 + m;
      int d = nt * 16 + ln;
      ao[(size_t)(b * NQ_ + q) * (H_ * DK_) + h * DK_ + d] =
          __builtin_bit_cast(unsigned short, (__bf16)(O[nt][i] * run_sum[i]));
    }
  }
}

// ---------------------------------------------------------------------------
// Kernel 3: output projection, attn_out(bf16) x Wo + bo -> f32.
// The bf16 A tile is staged by the TDM; Wo needs f32->bf16 so it stays manual
// and overlaps the TDM transfer.
// ---------------------------------------------------------------------------
__global__ __launch_bounds__(128) void oproj_kernel(
    const unsigned short* __restrict__ ao, const float* __restrict__ Wo,
    const float* __restrict__ bo, float* __restrict__ out) {
  __shared__ __bf16 sA[64][34];
  __shared__ __bf16 sBt[64][34];
  const int tid = threadIdx.x;
  const int wave = tid >> 5;
  const int lane = tid & 31;
  const int ln = lane & 15, lg = lane >> 4;
  const int col0 = blockIdx.x * 64;
  const int row0 = blockIdx.y * 64;

  v8f acc[4] = {};
  for (int k0 = 0; k0 < DM_; k0 += 32) {
#if USE_TDM
    if (wave == 0) {
      // 64 rows x 32 bf16; row = 64 B = 16 dwords, +1 dword pad -> code 3.
      tdm_load_2d_bf16((unsigned)(size_t)&sA[0][0],
                       ao + (size_t)row0 * DM_ + k0, 32, 64, DM_, 3);
    }
#else
    for (int t = tid; t < 64 * 32; t += 128) {
      int r = t >> 5, c = t & 31;
      sA[r][c] = __builtin_bit_cast(__bf16, ao[(size_t)(row0 + r) * DM_ + k0 + c]);
    }
#endif
    for (int t = tid; t < 32 * 64; t += 128) {
      int kk = t >> 6, c = t & 63;
      sBt[c][kk] = (__bf16)Wo[(size_t)(k0 + kk) * DM_ + col0 + c];
    }
#if USE_TDM
    if (wave == 0) __builtin_amdgcn_s_wait_tensorcnt(0);
#endif
    __syncthreads();
    v16bf a = a_frag(&sA[wave * 16 + ln][0], lg);
#pragma unroll
    for (int nt = 0; nt < 4; ++nt) {
      v16bf b = b_frag(&sBt[nt * 16 + ln][0], lg);
      acc[nt] = wmma_bf16(a, b, acc[nt]);
    }
    __syncthreads();
  }
#pragma unroll
  for (int nt = 0; nt < 4; ++nt)
#pragma unroll
    for (int i = 0; i < 8; ++i) {
      int m = lg * 8 + i;
      int n = col0 + nt * 16 + ln;
      out[(size_t)(row0 + wave * 16 + m) * DM_ + n] = acc[nt][i] + bo[n];
    }
}

extern "C" void kernel_launch(void* const* d_in, const int* in_sizes, int n_in,
                              void* d_out, int out_size, void* d_ws,
                              size_t ws_size, hipStream_t stream) {
  const float* queries = (const float*)d_in[0];
  const float* keys = (const float*)d_in[1];
  const float* values = (const float*)d_in[2];
  const unsigned char* mask = (const unsigned char*)d_in[3];
  const float* geo = (const float*)d_in[4];
  const float* Wq = (const float*)d_in[5];
  const float* bq = (const float*)d_in[6];
  const float* Wk = (const float*)d_in[7];
  const float* bk = (const float*)d_in[8];
  const float* Wv = (const float*)d_in[9];
  const float* bv = (const float*)d_in[10];
  const float* Wo = (const float*)d_in[11];
  const float* bo = (const float*)d_in[12];
  float* out = (float*)d_out;

  const size_t headElems = (size_t)B_ * H_ * NQ_ * DK_;  // 4 Mi bf16 each
  unsigned short* qh = (unsigned short*)d_ws;
  unsigned short* kh = qh + headElems;
  unsigned short* vh = kh + headElems;
  unsigned short* ao = vh + headElems;  // (b, nq, H*DK) bf16

  qkv_proj_kernel<<<dim3(DM_ / 64, (B_ * NQ_) / 64, 3), 128, 0, stream>>>(
      queries, keys, values, Wq, Wk, Wv, bq, bk, bv, qh, kh, vh);
  attn_kernel<<<dim3(NQ_ / 64, H_, B_), 128, 0, stream>>>(qh, kh, vh, geo, mask, ao);
  oproj_kernel<<<dim3(DM_ / 64, (B_ * NQ_) / 64), 128, 0, stream>>>(ao, Wo, bo, out);
}
